// NF4Linear_36232344109582
// MI455X (gfx1250) — matrix operand order repaired and checked by reading, not compiled, not run on hardware
//
#include <hip/hip_runtime.h>

typedef __attribute__((ext_vector_type(16))) _Float16 v16h;
typedef __attribute__((ext_vector_type(8)))  _Float16 v8h;
typedef __attribute__((ext_vector_type(2)))  _Float16 h2v;
typedef __attribute__((ext_vector_type(8)))  float    v8f;

typedef __attribute__((ext_vector_type(4))) unsigned int u32x4;
typedef __attribute__((ext_vector_type(8))) int          i32x8;
typedef __attribute__((ext_vector_type(4))) int          i32x4;

#ifndef __has_builtin
#define __has_builtin(x) 0
#endif

#if __has_builtin(__builtin_amdgcn_tensor_load_to_lds)
#define HAVE_TDM 1
#if __has_include(<hip/amd_detail/amd_gfx1250_TDM.h>)
#define TDM_LOAD(g0, g1, z4a, z4b, z8, cpol) \
  __builtin_amdgcn_tensor_load_to_lds((g0), (g1), (z4a), (z4b), (z8), (cpol))
#else
#define TDM_LOAD(g0, g1, z4a, z4b, z8, cpol) \
  __builtin_amdgcn_tensor_load_to_lds((g0), (g1), (z4a), (z4b), (cpol))
#endif
#else
#define HAVE_TDM 0
#endif

#define M_DIM 4096
#define K_DIM 4096
#define N_DIM 11008
#define GROUP 128

#define BM 128
#define BN 128
#define BK 32
#define KTILES (K_DIM / BK)     // 128
#define LDA 80                  // bytes per A row  (64B data + 16B TDM pad)
#define LDB 80                  // bytes per B col  (64B data + 16B pad)
#define ABYTES (BM * LDA)       // 10240
#define BBYTES (BN * LDB)       // 10240
#define BUFSEL 0x8000u          // buffer-1 offset (XOR toggle)
#define SMEM_BYTES (BUFSEL + ABYTES + BBYTES)   // 53248

// fp16 bit patterns of the 16 NF4 quantization levels
__device__ const unsigned short NF4_H[16] = {
  0xBC00, 0xB992, 0xB833, 0xB652, 0xB48D, 0xB1EA, 0xADD4, 0x0000,
  0x2D18, 0x3126, 0x33E0, 0x3568, 0x370D, 0x3880, 0x39C9, 0x3C00
};

__global__ __launch_bounds__(256)
void nf4_gemm_wmma(const _Float16* __restrict__ X,
                   const int*      __restrict__ Wq,
                   const _Float16* __restrict__ Sc,
                   _Float16*       __restrict__ C)
{
  __shared__ unsigned char smem[SMEM_BYTES];
  // buffer c (c in {0, BUFSEL}): A at smem+c, B at smem+c+ABYTES

  const int tid  = threadIdx.x;
  const int lane = tid & 31;
  const int wave = tid >> 5;
  const int wm   = wave & 3;    // M sub-block: wm*32
  const int wn   = wave >> 2;   // N sub-block: wn*64

  const int bm0 = blockIdx.y * BM;
  const int bn0 = blockIdx.x * BN;

  // ---- W staging map: 16 packed rows x 128 n dwords -> 8 dwords per thread ----
  const int br = tid >> 4;                       // packed row 0..15  (k = 2*br, 2*br+1)
  const int bn = (tid & 15) * 8;                 // 8 consecutive n
  const int* gW = Wq + (size_t)br * N_DIM + (bn0 + bn);

#if !HAVE_TDM
  const int am = tid >> 1;
  const int ak = (tid & 1) * 16;
  const _Float16* gA = X + (size_t)(bm0 + am) * K_DIM + ak;
#endif

  // NF4 LUT: lane i holds fp16 bits of level (i & 15); lookup = ds_bpermute
  int lut = (int)NF4_H[lane & 15];

  v8f acc[2][4];
  #pragma unroll
  for (int i = 0; i < 2; ++i)
    #pragma unroll
    for (int j = 0; j < 4; ++j)
      acc[i][j] = (v8f)0.0f;

  v8h scv = {};   // per-thread scales for its 8 n-columns

#if HAVE_TDM
  // ---- TDM descriptor, invariant parts (ISA cdna5 §8) ----
  // Tile 8 x 128 of 8B elements (64B/row), tensor row stride = 1024 elem;
  // pad_enable: +4 dwords LDS pad per 16 dwords -> LDS row stride 80B == LDA.
  const unsigned ldsA = (unsigned)(size_t)smem;  // low 32 bits = LDS offset
  i32x8 g1 = {};
  g1[0] = (3 << 16)      // data_size = 8B
        | (1 << 20)      // pad_enable
        | (3 << 22)      // pad_interval: 16 dwords
        | (3 << 25);     // pad_amount:   4 dwords
  g1[1] = (1024 << 16);  // tensor_dim0 = 1024 (low 16)
  g1[2] = (M_DIM << 16); // tensor_dim1 low16
  g1[3] = (8 << 16);     // tile_dim0 = 8
  g1[4] = 128;           // tile_dim1 = 128
  g1[5] = 1024;          // tensor_dim0_stride (low 32)
  const i32x4 z4 = {};
  const i32x8 z8 = {};
#endif

  // prologue: weight tile 0 into registers
  int4 wa = *(const int4*)(gW);
  int4 wb = *(const int4*)(gW + 4);

  unsigned boff = 0;   // current LDS buffer offset (0 or BUFSEL)

  for (int kt = 0; kt < KTILES; ++kt) {
    if ((kt & 3) == 0) {
      const int g = kt >> 2;
      scv = *(const v8h*)(Sc + (size_t)g * N_DIM + (bn0 + bn));
    }

#if HAVE_TDM
    // ---- A tile via Tensor Data Mover (one DMA per workgroup) ----
    if (wave == 0) {
      unsigned long long ga = (unsigned long long)(size_t)X
          + ((size_t)bm0 * K_DIM + (size_t)kt * BK) * 2;
      u32x4 g0;
      g0[0] = 1u;                                   // count = 1
      g0[1] = ldsA + boff;                          // lds_addr (current buffer)
      g0[2] = (unsigned)(ga & 0xFFFFFFFFu);         // global_addr[31:0]
      g0[3] = (unsigned)((ga >> 32) & 0x01FFFFFFu)  // global_addr[56:32]
            | (2u << 30);                           // type = 2
      TDM_LOAD(g0, g1, z4, z4, z8, 0);
    }
#else
    {
      uint4 a0 = *(const uint4*)(gA + (size_t)kt * BK);
      uint4 a1 = *(const uint4*)(gA + (size_t)kt * BK + 8);
      *(uint4*)(smem + boff + am * LDA + ak * 2)      = a0;
      *(uint4*)(smem + boff + am * LDA + ak * 2 + 16) = a1;
    }
#endif

    // ---- dequant tile kt (regs loaded last iteration) -> B buffer ----
    unsigned char* sB = smem + boff + ABYTES;
    const int wd[8] = { wa.x, wa.y, wa.z, wa.w, wb.x, wb.y, wb.z, wb.w };
    #pragma unroll
    for (int q = 0; q < 8; ++q) {
      const int d   = wd[q];
      const int vlo = __shfl(lut, d & 15, 32);          // k = 2*br
      const int vhi = __shfl(lut, (d >> 4) & 15, 32);   // k = 2*br+1
      unsigned int pair = (unsigned int)(vlo & 0xFFFF) | ((unsigned int)vhi << 16);
      h2v wp = __builtin_bit_cast(h2v, pair);
      h2v s2; s2.x = scv[q]; s2.y = scv[q];
      h2v r = wp * s2;                                  // v_pk_mul_f16
      *(h2v*)(sB + (bn + q) * LDB + br * 4) = r;        // ds_store_b32
    }

    // ---- issue next weight tile; latency covered by compute below ----
    {
      const int ktn = (kt + 1 < KTILES) ? (kt + 1) : kt;
      const int* gWn = gW + (size_t)ktn * 16 * N_DIM;
      wa = *(const int4*)(gWn);
      wb = *(const int4*)(gWn + 4);
      const int ktp = (kt + 2 < KTILES) ? (kt + 2) : kt;
      __builtin_prefetch(gW + (size_t)ktp * 16 * N_DIM, 0, 1);
    }

#if HAVE_TDM
    if (wave == 0) __builtin_amdgcn_s_wait_tensorcnt(0);
#endif
    __syncthreads();   // single barrier per k-tile (double-buffered)

    // ---- fragments from current buffer ----
    const unsigned char* sA = smem + boff;
    union { v16h v; v8h h[2]; } af[2], bf[4];
    #pragma unroll
    for (int i = 0; i < 2; ++i) {
      const int arow = wm * 32 + i * 16 + (lane & 15);
      const unsigned char* pa = sA + arow * LDA + (lane >> 4) * 16;
      af[i].h[0] = *(const v8h*)(pa);        // K 0..7   (lanes>=16: 8..15)
      af[i].h[1] = *(const v8h*)(pa + 32);   // K 16..23 (lanes>=16: 24..31)
    }
    #pragma unroll
    for (int j = 0; j < 4; ++j) {
      const int bcol = wn * 64 + j * 16 + (lane & 15);
      const unsigned char* pb = sB + bcol * LDB + (lane >> 4) * 16;
      bf[j].h[0] = *(const v8h*)(pb);
      bf[j].h[1] = *(const v8h*)(pb + 32);
    }

    // ---- 8 WMMAs per wave per k-step ----
    #pragma unroll
    for (int i = 0; i < 2; ++i)
      #pragma unroll
      for (int j = 0; j < 4; ++j)
        acc[i][j] = __builtin_amdgcn_wmma_f32_16x16x32_f16(
            false, af[i].v, false, bf[j].v,
            (short)0, acc[i][j], false, false);

    boff ^= BUFSEL;
  }

  // ---- epilogue: f32 -> fp16 store ----
  #pragma unroll
  for (int i = 0; i < 2; ++i) {
    #pragma unroll
    for (int j = 0; j < 4; ++j) {
      const int row0 = bm0 + wm * 32 + i * 16 + (lane >> 4) * 8;
      const int col  = bn0 + wn * 64 + j * 16 + (lane & 15);
      #pragma unroll
      for (int v = 0; v < 8; ++v)
        C[(size_t)(row0 + v) * N_DIM + col] = (_Float16)acc[i][j][v];
    }
  }
}

extern "C" void kernel_launch(void* const* d_in, const int* in_sizes, int n_in,
                              void* d_out, int out_size, void* d_ws, size_t ws_size,
                              hipStream_t stream) {
  (void)in_sizes; (void)n_in; (void)out_size; (void)d_ws; (void)ws_size;
  const _Float16* x  = (const _Float16*)d_in[0];
  const int*      wq = (const int*)d_in[1];
  const _Float16* sc = (const _Float16*)d_in[2];
  _Float16*       out = (_Float16*)d_out;

  dim3 grid(N_DIM / BN, M_DIM / BM);   // 86 x 32 workgroups
  nf4_gemm_wmma<<<grid, 256, 0, stream>>>(x, wq, sc, out);
}
